// RatSpn_21818433863920
// MI455X (gfx1250) — matrix operand order repaired and checked by reading, compiled, not run on hardware
//
#include <hip/hip_runtime.h>
#include <hip/hip_bf16.h>
#include <math.h>

typedef __attribute__((ext_vector_type(16))) _Float16 v16h;
typedef __attribute__((ext_vector_type(8)))  _Float16 v8h;
typedef __attribute__((ext_vector_type(8)))  float    v8f;

#define BATCH 128
#define LOG2PI_F 1.8378770664093453f

// ---------------------------------------------------------------------------
// Leaf kernel: lp0[b][r][g] = sum_v gaussian logpdf, b<128, r<64, g<32
// ---------------------------------------------------------------------------
__global__ void leaf_kernel(const float* __restrict__ x,
                            const float* __restrict__ means,
                            const float* __restrict__ sigp,
                            float* __restrict__ lp0) {
    int idx = blockIdx.x * blockDim.x + threadIdx.x;  // 0 .. 128*64*32-1
    if (idx >= BATCH * 64 * 32) return;
    int g = idx & 31;
    int r = (idx >> 5) & 63;
    int b = idx >> 11;
    float acc = 0.0f;
#pragma unroll
    for (int v = 0; v < 2; ++v) {
        float xv = x[b * 128 + r * 2 + v];
        int mi = (r * 2 + v) * 32 + g;
        float mu = means[mi];
        float sp = sigp[mi];
        float sig = 0.1f + 0.9f * (1.0f / (1.0f + expf(-sp)));   // min..max sigmoid
        float z = xv - mu;
        acc += -0.5f * z * z / sig - 0.5f * logf(sig) - 0.5f * LOG2PI_F;
    }
    lp0[idx] = acc;
}

// ---------------------------------------------------------------------------
// Weight softmax kernel: softmax over k (axis=1) of w[h][k][o], stored
// TRANSPOSED + f16:  wt[h][o][k], k contiguous (for WMMA B-fragment loads).
// One wave per (h, o).  o >= N (padding columns) are zero-filled.
// ---------------------------------------------------------------------------
__global__ void wsoftmax_kernel(const float* __restrict__ w,
                                _Float16* __restrict__ wt,
                                int N, int Npad) {
    int h = blockIdx.x / Npad;
    int o = blockIdx.x % Npad;
    int lane = threadIdx.x;  // 0..31
    _Float16* dst = wt + ((size_t)(h * Npad + o)) * 1024;
    if (o >= N) {
        for (int t = 0; t < 32; ++t) dst[t * 32 + lane] = (_Float16)0.0f;
        return;
    }
    const float* src = w + (size_t)h * 1024 * N + o;
    float vals[32];
    float m = -1e30f;
#pragma unroll
    for (int t = 0; t < 32; ++t) {
        float v = src[(size_t)(t * 32 + lane) * N];
        vals[t] = v;
        m = fmaxf(m, v);
    }
    for (int off = 16; off > 0; off >>= 1) m = fmaxf(m, __shfl_xor(m, off, 32));
    float s = 0.0f;
#pragma unroll
    for (int t = 0; t < 32; ++t) s += expf(vals[t] - m);
    for (int off = 16; off > 0; off >>= 1) s += __shfl_xor(s, off, 32);
    float inv = 1.0f / s;
#pragma unroll
    for (int t = 0; t < 32; ++t)
        dst[t * 32 + lane] = (_Float16)(expf(vals[t] - m) * inv);
}

// ---------------------------------------------------------------------------
// Layer kernel: one workgroup per output region h.
//   lp_out[b][h][o] = M[b] + log( P(128x1024) @ Wt(1024xNpad) )
// where P[b, ks*32+kl] = a2[b][ks] * a1[b][kl]   (rank-1 per K-tile).
// 16 waves: wave = (m_tile 0..7) x (n_tile 0..1).
// K reduction split into TWO independent accumulator chains (even/odd ks)
// so the scheduler can fill WMMA->VALU co-exec hazard slots with the other
// chain's fragment-scaling pk_muls instead of v_nops.
// ---------------------------------------------------------------------------
__global__ __launch_bounds__(512) void layer_kernel(
    const float* __restrict__ lpin, int R_in,
    const _Float16* __restrict__ wt, int Npad,
    float* __restrict__ lpout, int hr /*half regions*/, int Nvalid,
    int finalFlag) {
    int h = blockIdx.x;
    __shared__ alignas(16) _Float16 a1sh[BATCH][32];
    __shared__ alignas(16) _Float16 a2sh[BATCH][32];
    __shared__ float Mv[BATCH];
    __shared__ float mtmp[2][BATCH];

    int tid = threadIdx.x;

    // -------- Phase 1: per-batch max + exp of the two child regions --------
    if (tid < 256) {
        int b = tid & 127;
        int which = tid >> 7;  // 0 -> lp1 (even region), 1 -> lp2 (odd region)
        const float* row = lpin + ((size_t)b * R_in + (2 * h + which)) * 32;
        float v[32];
        float m = -1e30f;
#pragma unroll
        for (int j = 0; j < 32; ++j) {
            v[j] = row[j];
            m = fmaxf(m, v[j]);
        }
        _Float16* dstA = which ? a2sh[b] : a1sh[b];
#pragma unroll
        for (int j = 0; j < 32; ++j) dstA[j] = (_Float16)expf(v[j] - m);
        mtmp[which][b] = m;
    }
    __syncthreads();
    if (tid < BATCH) Mv[tid] = mtmp[0][tid] + mtmp[1][tid];
    __syncthreads();

    // -------- Phase 2: WMMA GEMM  S = P @ Wt --------
    int wv = tid >> 5;
    int lane = tid & 31;
    int mt = wv >> 1;   // 0..7  -> rows b = mt*16 .. mt*16+15
    int nt = wv & 1;    // 0..1  -> cols o = nt*16 .. nt*16+15
    if (nt * 16 >= Npad) return;  // uniform per wave (only last layer, Npad=16)

    int mrow = lane & 15;
    int b = mt * 16 + mrow;        // A-matrix row (batch index) for this lane
    int o = nt * 16 + mrow;        // B-matrix column for this lane
    int k0 = (lane < 16) ? 0 : 16; // A fragment K base (contiguous K, matches B)
    int kb = (lane < 16) ? 0 : 16; // B fragment K base (32x16 layout)

    const _Float16* bcol = wt + ((size_t)(h * Npad + o)) * 1024 + kb;

    // a1 pieces are invariant across ks: hoist the LDS loads out of the loop.
    v8h lo = *(const v8h*)&a1sh[b][k0];        // K = k0..k0+7
    v8h hi = *(const v8h*)&a1sh[b][k0 + 8];    // K = k0+8..k0+15

    v8f c0 = {};
    v8f c1 = {};
#pragma unroll
    for (int ks = 0; ks < 32; ks += 2) {
        // ---- chain 0: even ks ----
        {
            _Float16 s = a2sh[b][ks];
            v16h af;
#pragma unroll
            for (int e = 0; e < 8; ++e) {
                af[e]     = s * lo[e];
                af[e + 8] = s * hi[e];
            }
            v16h bf = *(const v16h*)(bcol + ks * 32);
            c0 = __builtin_amdgcn_wmma_f32_16x16x32_f16(
                false, af, false, bf, (short)0, c0, false, false);
        }
        // ---- chain 1: odd ks (independent of chain 0) ----
        {
            _Float16 s = a2sh[b][ks + 1];
            v16h af;
#pragma unroll
            for (int e = 0; e < 8; ++e) {
                af[e]     = s * lo[e];
                af[e + 8] = s * hi[e];
            }
            v16h bf = *(const v16h*)(bcol + (ks + 1) * 32);
            c1 = __builtin_amdgcn_wmma_f32_16x16x32_f16(
                false, af, false, bf, (short)0, c1, false, false);
        }
    }

    // -------- Store: lp = M + log(S), per C/D 16x16 f32 layout --------
    int rbase = mt * 16 + ((lane < 16) ? 0 : 8);
#pragma unroll
    for (int g = 0; g < 8; ++g) {
        int br = rbase + g;
        float val = Mv[br] + logf(c0[g] + c1[g]);
        if (finalFlag) {
            if (o < Nvalid) lpout[(size_t)br * Nvalid + o] = val;
        } else {
            lpout[((size_t)br * hr + h) * 32 + o] = val;
        }
    }
}

// ---------------------------------------------------------------------------
// Host launcher
// ---------------------------------------------------------------------------
extern "C" void kernel_launch(void* const* d_in, const int* in_sizes, int n_in,
                              void* d_out, int out_size, void* d_ws, size_t ws_size,
                              hipStream_t stream) {
    const float* x     = (const float*)d_in[0];
    const float* means = (const float*)d_in[1];
    const float* sigp  = (const float*)d_in[2];
    const float* w[6];
    for (int l = 0; l < 6; ++l) w[l] = (const float*)d_in[3 + l];
    float* out = (float*)d_out;

    char* ws = (char*)d_ws;
    float* lpA = (float*)(ws);                  // 128*64*32 f32 = 1 MB
    float* lpB = (float*)(ws + 1048576);        // 128*32*32 f32 = 512 KB

    static const int    Rl[6]   = {32, 16, 8, 4, 2, 1};     // regions per w
    static const int    Nl[6]   = {32, 32, 32, 32, 32, 10}; // valid outputs
    static const int    Npad[6] = {32, 32, 32, 32, 32, 16}; // padded outputs
    static const size_t wbytes[6] = {2097152, 1048576, 524288, 262144, 131072, 32768};

    _Float16* wt[6];
    size_t off = 1572864;
    for (int l = 0; l < 6; ++l) {
        wt[l] = (_Float16*)(ws + off);
        off += wbytes[l];
    }

    // 1) Gaussian leaves
    leaf_kernel<<<1024, 256, 0, stream>>>(x, means, sigp, lpA);

    // 2) Weight softmax (f32 -> normalized f16, transposed [h][o][k])
    for (int l = 0; l < 6; ++l)
        wsoftmax_kernel<<<Rl[l] * Npad[l], 32, 0, stream>>>(w[l], wt[l], Nl[l], Npad[l]);

    // 3) Six sum-product layers (ping-pong lpA/lpB, final -> d_out)
    int R_in = 64;
    for (int l = 0; l < 6; ++l) {
        int hr = R_in / 2;
        const float* src = (l % 2 == 0) ? lpA : lpB;
        float* dst = (l == 5) ? out : ((l % 2 == 0) ? lpB : lpA);
        layer_kernel<<<hr, 512, 0, stream>>>(src, R_in, wt[l], Npad[l], dst, hr,
                                             Nl[l], (l == 5) ? 1 : 0);
        R_in = hr;
    }
}
